// MultiheadAttention_73933567034173
// MI455X (gfx1250) — matrix operand order repaired and checked by reading, not compiled, not run on hardware
//
#include <hip/hip_runtime.h>
#include <cstdint>

// ---------------- types ----------------
typedef __bf16 v16bf __attribute__((ext_vector_type(16)));
typedef __bf16 v2bf  __attribute__((ext_vector_type(2)));
typedef float  v8f   __attribute__((ext_vector_type(8)));

struct alignas(16) U4 { unsigned x, y, z, w; };
struct alignas(16) F4 { float x, y, z, w; };
struct alignas(8)  F2 { float x, y; };

union FragB {
    v16bf v;
    unsigned short h[16];
    unsigned u[8];
    U4 q[2];
};

// fp32 -> bf16 via native fptrunc (pattern-matches v_cvt_pk_bf16_f32 when present)
static __device__ __forceinline__ unsigned short bfu(float f) {
    return __builtin_bit_cast(unsigned short, (__bf16)f);
}
#if __has_builtin(__builtin_amdgcn_cvt_pk_bf16_f32)
static __device__ __forceinline__ unsigned pk2(float a, float b) {
    auto r = __builtin_amdgcn_cvt_pk_bf16_f32(a, b);   // v_cvt_pk_bf16_f32
    union Cvt { decltype(r) r2; unsigned u; } x;
    x.r2 = r;
    return x.u;
}
#else
static __device__ __forceinline__ unsigned pk2(float a, float b) {
    v2bf t;
    t.x = (__bf16)a;
    t.y = (__bf16)b;
    return __builtin_bit_cast(unsigned, t);
}
#endif

// 16 contiguous fp32 -> one bf16 B-fragment (K = half*16 + j)
static __device__ __forceinline__ void packB16(FragB& b, const float* p) {
    const F4* q = (const F4*)p;
    F4 f0 = q[0], f1 = q[1], f2 = q[2], f3 = q[3];
    b.u[0] = pk2(f0.x, f0.y); b.u[1] = pk2(f0.z, f0.w);
    b.u[2] = pk2(f1.x, f1.y); b.u[3] = pk2(f1.z, f1.w);
    b.u[4] = pk2(f2.x, f2.y); b.u[5] = pk2(f2.z, f2.w);
    b.u[6] = pk2(f3.x, f3.y); b.u[7] = pk2(f3.z, f3.w);
}
// fp32 row (32-float K window) -> bf16 A-fragment (K = hf*8+j and 16+hf*8+j)
static __device__ __forceinline__ void packA32(FragB& a, const float* row, int hf) {
    const F4* q0 = (const F4*)(row + hf * 8);
    const F4* q1 = (const F4*)(row + 16 + hf * 8);
    F4 f0 = q0[0], f1 = q0[1], f2 = q1[0], f3 = q1[1];
    a.u[0] = pk2(f0.x, f0.y); a.u[1] = pk2(f0.z, f0.w);
    a.u[2] = pk2(f1.x, f1.y); a.u[3] = pk2(f1.z, f1.w);
    a.u[4] = pk2(f2.x, f2.y); a.u[5] = pk2(f2.z, f2.w);
    a.u[6] = pk2(f3.x, f3.y); a.u[7] = pk2(f3.z, f3.w);
}
static __device__ __forceinline__ v8f wmma_bf16(const FragB& a, const FragB& b, v8f c) {
    return __builtin_amdgcn_wmma_f32_16x16x32_bf16(false, a.v, false, b.v,
                                                   (short)0, c, false, false);
}
static __device__ __forceinline__ float fast_exp(float x) {
#if __has_builtin(__builtin_amdgcn_exp2f)
    return __builtin_amdgcn_exp2f(x * 1.44269504088896f);
#else
    return exp2f(x * 1.44269504088896f);
#endif
}
static __device__ __forceinline__ float fast_rcp(float x) {
#if __has_builtin(__builtin_amdgcn_rcpf)
    return __builtin_amdgcn_rcpf(x);
#else
    return 1.0f / x;
#endif
}

// ------- CDNA5 async global->LDS (Tensor/async data path, §08) -------
#define USE_ASYNC_LDS 1
static __device__ __forceinline__ void async_cp16(void* lds, const void* g) {
#if USE_ASYNC_LDS
    unsigned loff = (unsigned)(size_t)lds;   // low 32 bits = LDS byte offset
    asm volatile("global_load_async_to_lds_b128 %0, %1, off"
                 :: "v"(loff), "v"(g) : "memory");
#else
    *(F4*)lds = *(const F4*)g;
#endif
}
static __device__ __forceinline__ void async_wait() {
#if USE_ASYNC_LDS
    asm volatile("s_wait_asynccnt 0x0" ::: "memory");
#endif
}

// ---------------- problem constants ----------------
#define LSEQ 1024   // L == S
#define BTOT 64     // N*H
#define DH   64
#define LC   64     // l-chunk
#define NCH  16     // 1024 / LC

// ---------------- workspace layout (bytes) ----------------
#define WS_WQ   ((size_t)0)
#define WS_WK   (WS_WQ + 512u * 512u * 2u)
#define WS_WV   (WS_WK + 512u * 512u * 2u)
#define WS_WO   (WS_WV + 512u * 512u * 2u)
#define WS_QB   (WS_WO + 512u * 512u * 2u)                    // [b][l][d] bf16
#define WS_KB   (WS_QB + (size_t)BTOT * LSEQ * DH * 2u)       // [b][s][d] bf16
#define WS_VT   (WS_KB + (size_t)BTOT * LSEQ * DH * 2u)       // [b][d][s] bf16
#define WS_CTX  (WS_VT + (size_t)BTOT * LSEQ * DH * 2u)       // [b][l][d] f32
#define WS_PS   (WS_CTX + (size_t)BTOT * LSEQ * DH * 4u)      // [b][lc][s] f32
#define WS_WBUF (WS_PS + (size_t)BTOT * LC * LSEQ * 4u)       // [b][lc][s] f32
#define WS_END  (WS_WBUF + (size_t)BTOT * LC * LSEQ * 4u)

// =============== fp32 -> bf16 elementwise ===============
__global__ void k_cvt_bf16(const float* __restrict__ src,
                           unsigned* __restrict__ dst, int npairs) {
    int i = blockIdx.x * 256 + threadIdx.x;
    if (i < npairs) {
        F2 f = *(const F2*)(src + (size_t)i * 2);
        dst[i] = pk2(f.x, f.y);
    }
}

// =============== input projection GEMM ===============
// X (8192 x 512 fp32, row = l*8+n) @ W^T(bf16) + bias; scatter to per-head bf16.
// grid (512, 4), block 256.  transposeV: store [b][d][t] instead of [b][t][d].
__global__ void k_proj(const float* __restrict__ X,
                       const unsigned short* __restrict__ Wbf,
                       const float* __restrict__ bias,
                       unsigned short* __restrict__ dst,
                       float scale, int transposeV) {
    __shared__ unsigned short As[2][16 * 40];   // double-buffered 16x32 bf16 tile
    const int m0 = blockIdx.x * 16, n0 = blockIdx.y * 128;
    const int w = threadIdx.x >> 5, lane = threadIdx.x & 31;
    const int ln = lane & 15, hf = lane >> 4;
    const int nw = n0 + w * 16;
    const int sr = threadIdx.x >> 4, sc0 = (threadIdx.x * 2) & 31;
    const float* xrow = X + (size_t)(m0 + sr) * 512 + sc0;
    F2 f = *(const F2*)xrow;                    // k-step 0 preload
    v8f acc = {};
    for (int kk = 0; kk < 16; ++kk) {
        const int buf = kk & 1;
        *(unsigned*)&As[buf][sr * 40 + sc0] = pk2(f.x, f.y);
        __syncthreads();
        if (kk + 1 < 16) f = *(const F2*)(xrow + (kk + 1) * 32);  // overlap next stage
        FragB a, b;
        a.q[0] = *(const U4*)&As[buf][ln * 40 + hf * 8];
        a.q[1] = *(const U4*)&As[buf][ln * 40 + 16 + hf * 8];
        const unsigned short* wr = Wbf + (size_t)(nw + ln) * 512 + kk * 32 + hf * 8;
        b.q[0] = *(const U4*)wr;
        b.q[1] = *(const U4*)(wr + 16);
        acc = wmma_bf16(a, b, acc);
    }
#pragma unroll
    for (int r = 0; r < 8; ++r) {
        int m = m0 + r + hf * 8;
        int l = m >> 3, nb = m & 7;
        int eo = nw + ln;
        int hh = eo >> 6, d = eo & 63;
        int b = nb * 8 + hh;
        unsigned short v = bfu((acc[r] + bias[eo]) * scale);
        if (transposeV) dst[((size_t)b * 64 + d) * 1024 + l] = v;
        else            dst[((size_t)b * 1024 + l) * 64 + d] = v;
    }
}

// =============== P1: positional scores ===============
// ps[b][l-lbase][s] = sum_d q[b,l,d]*pos_k[l,s,d].  grid (64, 4), block 256.
__global__ void k_pos_scores(const float* __restrict__ pos_k,
                             const unsigned short* __restrict__ qb,
                             float* __restrict__ ps, int lbase) {
    const int li = blockIdx.x, l = lbase + li;
    const int w = threadIdx.x >> 5, lane = threadIdx.x & 31;
    const int ln = lane & 15, hf = lane >> 4;
    const int btile = w & 3, sh = w >> 2;
    const int sbase = blockIdx.y * 256 + sh * 128;
    // A: all-head Q at row l (rows = b)
    const unsigned short* qr = qb + ((size_t)(btile * 16 + ln) * 1024 + l) * 64;
    FragB a0, a1;
    a0.q[0] = *(const U4*)(qr + hf * 16);       a0.q[1] = *(const U4*)(qr + hf * 16 + 8);
    a1.q[0] = *(const U4*)(qr + 32 + hf * 16);  a1.q[1] = *(const U4*)(qr + 40 + hf * 16);
    for (int st = 0; st < 8; ++st) {
        const int s0 = sbase + st * 16;
        const float* pk = pos_k + ((size_t)l * 1024 + s0 + ln) * 64;
        if (st + 1 < 8) __builtin_prefetch(pk + 16 * 64, 0, 1);
        FragB b0, b1;
        packB16(b0, pk + hf * 16);        // d = hf*16 + j
        packB16(b1, pk + 32 + hf * 16);   // d = 32 + hf*16 + j
        v8f acc = {};
        acc = wmma_bf16(a0, b0, acc);
        acc = wmma_bf16(a1, b1, acc);
#pragma unroll
        for (int r = 0; r < 8; ++r) {
            int b = btile * 16 + r + hf * 8;
            ps[((size_t)(b * 64 + li)) * 1024 + s0 + ln] = acc[r];
        }
    }
}

// =============== P2: content scores + softmax + w@V ===============
// grid (4, 64): x = l-tile within chunk, y = head b.  block 256.
__global__ void k_content_softmax(const unsigned short* __restrict__ qb,
                                  const unsigned short* __restrict__ kb,
                                  const float* __restrict__ ps,
                                  const unsigned short* __restrict__ vT,
                                  float* __restrict__ wbuf,
                                  float* __restrict__ ctx, int lbase) {
    __shared__ float sc[16 * 1028];
    __shared__ float red[2 * 16 * 17];
    __shared__ float part[4 * 256];
    const int lt = blockIdx.x * 16;          // chunk-local l offset
    const int b  = blockIdx.y;
    const int lglob0 = lbase + lt;
    const int w = threadIdx.x >> 5, lane = threadIdx.x & 31;
    const int ln = lane & 15, hf = lane >> 4;

    // ---- phase 1: content scores + add positional scores ----
    {
        const unsigned short* qr = qb + ((size_t)b * 1024 + lglob0 + ln) * 64;
        FragB a0, a1;
        a0.q[0] = *(const U4*)(qr + hf * 8);       a0.q[1] = *(const U4*)(qr + 16 + hf * 8);
        a1.q[0] = *(const U4*)(qr + 32 + hf * 8);  a1.q[1] = *(const U4*)(qr + 48 + hf * 8);
        const int sbase = w * 128;
        for (int st = 0; st < 8; ++st) {
            const int s0 = sbase + st * 16;
            const unsigned short* kr = kb + ((size_t)b * 1024 + s0 + ln) * 64;
            FragB b0, b1;
            b0.q[0] = *(const U4*)(kr + hf * 8);       b0.q[1] = *(const U4*)(kr + 16 + hf * 8);
            b1.q[0] = *(const U4*)(kr + 32 + hf * 8);  b1.q[1] = *(const U4*)(kr + 48 + hf * 8);
            v8f acc = {};
            acc = wmma_bf16(a0, b0, acc);
            acc = wmma_bf16(a1, b1, acc);
#pragma unroll
            for (int r = 0; r < 8; ++r) {
                int lrow = r + hf * 8, s = s0 + ln;
                float v = acc[r] + ps[((size_t)(b * 64 + lt + lrow)) * 1024 + s];
                v = fmaxf(v, -3.402823466e38f);   // clamp to finfo.min
                sc[lrow * 1028 + s] = v;
            }
        }
    }
    __syncthreads();

    // ---- phase 2: softmax over s, write w ----
    {
        const int r = threadIdx.x >> 4, seg = threadIdx.x & 15;
        float mx = -3.402823466e38f;
        for (int j = 0; j < 64; ++j) mx = fmaxf(mx, sc[r * 1028 + j * 16 + seg]);
        red[r * 17 + seg] = mx;
        __syncthreads();
        float M = -3.402823466e38f;
        for (int i = 0; i < 16; ++i) M = fmaxf(M, red[r * 17 + i]);
        float sum = 0.f;
        for (int j = 0; j < 64; ++j) {
            int idx = r * 1028 + j * 16 + seg;
            float e = fast_exp(sc[idx] - M);
            sc[idx] = e;
            sum += e;
        }
        red[16 * 17 + r * 17 + seg] = sum;
        __syncthreads();
        float T = 0.f;
        for (int i = 0; i < 16; ++i) T += red[16 * 17 + r * 17 + i];
        float inv = fast_rcp(T);
        for (int j = 0; j < 64; ++j) {
            int s = j * 16 + seg;
            float wv = sc[r * 1028 + s] * inv;
            sc[r * 1028 + s] = wv;
            wbuf[((size_t)(b * 64 + lt + r)) * 1024 + s] = wv;
        }
    }
    __syncthreads();

    // ---- phase 3: ctx = w @ V  (V stored d-major: direct B^T fragments) ----
    {
        const int dt = w & 3, kh = w >> 2, d0 = dt * 16;
        v8f acc = {};
        for (int kk = 0; kk < 16; ++kk) {
            const int k0 = kh * 512 + kk * 32;
            FragB a, bb;
            packA32(a, &sc[ln * 1028 + k0], hf);
            const unsigned short* vr = vT + ((size_t)b * 64 + d0 + ln) * 1024 + k0 + hf * 8;
            bb.q[0] = *(const U4*)vr;
            bb.q[1] = *(const U4*)(vr + 16);
            acc = wmma_bf16(a, bb, acc);
        }
        if (kh == 1) {
#pragma unroll
            for (int r = 0; r < 8; ++r) part[dt * 256 + lane * 8 + r] = acc[r];
        }
        __syncthreads();
        if (kh == 0) {
#pragma unroll
            for (int r = 0; r < 8; ++r) {
                float v = acc[r] + part[dt * 256 + lane * 8 + r];
                int m = r + hf * 8;
                ctx[((size_t)b * 1024 + lglob0 + m) * 64 + d0 + ln] = v;
            }
        }
    }
}

// =============== P3: ctx += w @ pos_v[l] ===============
// grid (64): one l per workgroup, all 64 heads.  block 256.
// pos_v staged global->LDS with CDNA5 async copies, double-buffered.
__global__ void k_posv_ctx(const float* __restrict__ pos_v,
                           const float* __restrict__ wbuf,
                           float* __restrict__ ctx, int lbase) {
    __shared__ float pvt[2][32 * 68];   // 32 s x 64 d, padded, double-buffered
    const int lc = blockIdx.x, l = lbase + lc;
    const int w = threadIdx.x >> 5, lane = threadIdx.x & 31;
    const int ln = lane & 15, hf = lane >> 4;
    const int btile = w & 3, dpair = w >> 2;
    const int sr = threadIdx.x >> 3, db = (threadIdx.x & 7) * 8;
    const float* srcbase = pos_v + ((size_t)l * 1024 + sr) * 64 + db;

    // prologue: async-stage step 0
    async_cp16(&pvt[0][sr * 68 + db],     srcbase);
    async_cp16(&pvt[0][sr * 68 + db + 4], srcbase + 4);
    async_wait();
    __syncthreads();

    v8f acc[2] = {{}, {}};
    for (int step = 0; step < 32; ++step) {
        const int buf = step & 1;
        if (step + 1 < 32) {   // issue next tile's async copies, overlap with WMMA
            const float* s2 = srcbase + (size_t)(step + 1) * 32 * 64;
            async_cp16(&pvt[buf ^ 1][sr * 68 + db],     s2);
            async_cp16(&pvt[buf ^ 1][sr * 68 + db + 4], s2 + 4);
        }
        const int s0 = step * 32;
        FragB a;  // rows = b, K = s
        packA32(a, wbuf + ((size_t)((btile * 16 + ln) * 64 + lc)) * 1024 + s0, hf);
#pragma unroll
        for (int i = 0; i < 2; ++i) {
            const int dcol = (dpair * 2 + i) * 16 + ln;
            FragB bb;
#pragma unroll
            for (int j = 0; j < 16; ++j)
                bb.h[j] = bfu(pvt[buf][(hf * 8 + j + (j & 8)) * 68 + dcol]);  // transpose via LDS
            acc[i] = wmma_bf16(a, bb, acc[i]);
        }
        async_wait();      // my copies into buf^1 are complete
        __syncthreads();   // everyone done reading buf / writing buf^1
    }
#pragma unroll
    for (int i = 0; i < 2; ++i) {
        const int d0 = (dpair * 2 + i) * 16;
#pragma unroll
        for (int r = 0; r < 8; ++r) {
            int b = btile * 16 + r + hf * 8;
            size_t ad = ((size_t)b * 1024 + l) * 64 + d0 + ln;
            ctx[ad] = ctx[ad] + acc[i][r];
        }
    }
}

// =============== P4: head-averaged attention weights ===============
__global__ void k_avg(const float* __restrict__ wbuf, float* __restrict__ avg, int lbase) {
    int idx = blockIdx.x * 256 + threadIdx.x;  // 8*64*1024 threads
    int s = idx & 1023, lc = (idx >> 10) & 63, n = idx >> 16;
    float sum = 0.f;
#pragma unroll
    for (int h = 0; h < 8; ++h)
        sum += wbuf[((size_t)((n * 8 + h) * 64 + lc)) * 1024 + s];
    avg[((size_t)n * 1024 + lbase + lc) * 1024 + s] = sum * 0.125f;
}

// =============== output projection ===============
// attn[l,n,eo] = sum_e ctx-gathered[l,n,e] * Wo[eo,e] + bo[eo].  grid (512, 4).
__global__ void k_out_proj(const float* __restrict__ ctx,
                           const unsigned short* __restrict__ Wob,
                           const float* __restrict__ bo,
                           float* __restrict__ out) {
    __shared__ unsigned short As[2][16 * 40];
    const int m0 = blockIdx.x * 16, n0 = blockIdx.y * 128;
    const int w = threadIdx.x >> 5, lane = threadIdx.x & 31;
    const int ln = lane & 15, hf = lane >> 4;
    const int nw = n0 + w * 16;
    const int sr = threadIdx.x >> 4, sc0 = (threadIdx.x * 2) & 31;
    const int m = m0 + sr, l = m >> 3, nb = m & 7;
    auto gload = [&](int kkk) -> F2 {  // gather 2 fp32 of logical (l,n,E) row from ctx
        int e = kkk * 32 + sc0;
        int bsrc = nb * 8 + (e >> 6), d = e & 63;
        return *(const F2*)(ctx + ((size_t)bsrc * 1024 + l) * 64 + d);
    };
    F2 f = gload(0);
    v8f acc = {};
    for (int kk = 0; kk < 16; ++kk) {
        const int buf = kk & 1;
        *(unsigned*)&As[buf][sr * 40 + sc0] = pk2(f.x, f.y);
        __syncthreads();
        if (kk + 1 < 16) f = gload(kk + 1);
        FragB a, b;
        a.q[0] = *(const U4*)&As[buf][ln * 40 + hf * 8];
        a.q[1] = *(const U4*)&As[buf][ln * 40 + 16 + hf * 8];
        const unsigned short* wr = Wob + (size_t)(nw + ln) * 512 + kk * 32 + hf * 8;
        b.q[0] = *(const U4*)wr;
        b.q[1] = *(const U4*)(wr + 16);
        acc = wmma_bf16(a, b, acc);
    }
#pragma unroll
    for (int r = 0; r < 8; ++r) {
        int mm = m0 + r + hf * 8;
        int eo = nw + ln;
        out[(size_t)mm * 512 + eo] = acc[r] + bo[eo];
    }
}

// =============== host-side orchestration ===============
extern "C" void kernel_launch(void* const* d_in, const int* in_sizes, int n_in,
                              void* d_out, int out_size, void* d_ws, size_t ws_size,
                              hipStream_t stream) {
    (void)in_sizes; (void)n_in; (void)out_size;
    if (ws_size < WS_END) return;   // workspace too small to run

    const float* query = (const float*)d_in[0];
    const float* key   = (const float*)d_in[1];
    const float* value = (const float*)d_in[2];
    const float* pos_k = (const float*)d_in[3];
    const float* pos_v = (const float*)d_in[4];
    const float* Wq = (const float*)d_in[5];  const float* bq = (const float*)d_in[6];
    const float* Wk = (const float*)d_in[7];  const float* bk = (const float*)d_in[8];
    const float* Wv = (const float*)d_in[9];  const float* bv = (const float*)d_in[10];
    const float* Wo = (const float*)d_in[11]; const float* bo = (const float*)d_in[12];

    unsigned char* ws = (unsigned char*)d_ws;
    unsigned short* wqb  = (unsigned short*)(ws + WS_WQ);
    unsigned short* wkb  = (unsigned short*)(ws + WS_WK);
    unsigned short* wvb  = (unsigned short*)(ws + WS_WV);
    unsigned short* wob  = (unsigned short*)(ws + WS_WO);
    unsigned short* qb   = (unsigned short*)(ws + WS_QB);
    unsigned short* kb   = (unsigned short*)(ws + WS_KB);
    unsigned short* vT   = (unsigned short*)(ws + WS_VT);
    float* ctx  = (float*)(ws + WS_CTX);
    float* ps   = (float*)(ws + WS_PS);
    float* wbuf = (float*)(ws + WS_WBUF);

    float* attn = (float*)d_out;                 // (L, N, E)
    float* avg  = attn + (size_t)1024 * 8 * 512; // (N, L, S)

    // weights -> bf16 (pairwise packed)
    const int npairs = 512 * 512 / 2;
    k_cvt_bf16<<<(npairs + 255) / 256, 256, 0, stream>>>(Wq, (unsigned*)wqb, npairs);
    k_cvt_bf16<<<(npairs + 255) / 256, 256, 0, stream>>>(Wk, (unsigned*)wkb, npairs);
    k_cvt_bf16<<<(npairs + 255) / 256, 256, 0, stream>>>(Wv, (unsigned*)wvb, npairs);
    k_cvt_bf16<<<(npairs + 255) / 256, 256, 0, stream>>>(Wo, (unsigned*)wob, npairs);

    // projections (q scaled by Dh^-0.5 = 0.125; V stored d-major)
    k_proj<<<dim3(512, 4), 256, 0, stream>>>(query, wqb, bq, qb, 0.125f, 0);
    k_proj<<<dim3(512, 4), 256, 0, stream>>>(key,   wkb, bk, kb, 1.0f,   0);
    k_proj<<<dim3(512, 4), 256, 0, stream>>>(value, wvb, bv, vT, 1.0f,   1);

    // attention core, chunked over l
    for (int c = 0; c < NCH; ++c) {
        const int lbase = c * LC;
        k_pos_scores<<<dim3(LC, 4), 256, 0, stream>>>(pos_k, qb, ps, lbase);
        k_content_softmax<<<dim3(4, BTOT), 256, 0, stream>>>(qb, kb, ps, vT, wbuf, ctx, lbase);
        k_posv_ctx<<<dim3(LC), 256, 0, stream>>>(pos_v, wbuf, ctx, lbase);
        k_avg<<<2048, 256, 0, stream>>>(wbuf, avg, lbase);
    }

    // final output projection
    k_out_proj<<<dim3(512, 4), 256, 0, stream>>>(ctx, wob, bo, attn);
}
